// SwinCrossAttentionBlock_79207786873484
// MI455X (gfx1250) — matrix operand order repaired and checked by reading, not compiled, not run on hardware
//
#include <hip/hip_runtime.h>
#include <hip/hip_bf16.h>
#include <math.h>

// ---------------------------------------------------------------------------
// Types & constants
// ---------------------------------------------------------------------------
typedef __bf16 bf16;
typedef __attribute__((ext_vector_type(16))) __bf16 v16bf;
typedef __attribute__((ext_vector_type(8)))  __bf16 v8bf;
typedef __attribute__((ext_vector_type(4)))  __bf16 v4bf;
typedef __attribute__((ext_vector_type(8)))  float  v8f;

constexpr int   kDIM   = 384;
constexpr int   kNH    = 12;
constexpr int   kHD    = 32;
constexpr int   kNTOK  = 50176;     // 4*112*112 == 1024 windows * 49 tokens
constexpr int   kHID   = 1536;
constexpr float kSCALE = 0.17677669529663687f;   // 32^-0.5

// ---------------------------------------------------------------------------
// WMMA fragment helpers (CDNA5 16x16x32 bf16, wave32), vectorized loads.
// A fragment: lane(li) = row M; elements 0..15 -> K = 16*half.. : one 32B run.
// B fragment: lane(li) = col N; elements 0..15 -> K = 16*half.. : one 32B run.
// C fragment: element r -> M = r + 8*half, N = lane&15
// ---------------------------------------------------------------------------
__device__ __forceinline__ v16bf frag_a(const bf16* base, int sL) {
  const int lane = threadIdx.x & 31;
  const int li = lane & 15, half = lane >> 4;
  return *(const v16bf*)(base + li * sL + 16 * half);
}

__device__ __forceinline__ v16bf frag_b(const bf16* base, int sL) {
  const int lane = threadIdx.x & 31;
  const int li = lane & 15, half = lane >> 4;
  return *(const v16bf*)(base + li * sL + 16 * half);
}

__device__ __forceinline__ v8f wmma_bf16(v16bf a, v16bf b, v8f c) {
  return __builtin_amdgcn_wmma_f32_16x16x32_bf16(false, a, false, b, (short)0, c,
                                                 false, false);
}

// ---------------------------------------------------------------------------
// fp32 -> bf16 weight conversion (vectorized, n % 4 == 0 for all weights)
// ---------------------------------------------------------------------------
__global__ void cvt_f32_bf16(const float* __restrict__ in, bf16* __restrict__ out,
                             int n4) {
  int i = blockIdx.x * blockDim.x + threadIdx.x;
  if (i < n4) {
    const float4 v = ((const float4*)in)[i];
    v4bf o;
    o[0] = (bf16)v.x; o[1] = (bf16)v.y; o[2] = (bf16)v.z; o[3] = (bf16)v.w;
    ((v4bf*)out)[i] = o;
  }
}

// ---------------------------------------------------------------------------
// Precompute padded bias+mask table: BM[wim][head][64][64]
//   valid (row,col < 49): rpb[rpi[row,col], head] + mask[wim,row,col]
//   padding: -1e30 (kills padded cols in softmax; padded rows neutralized later)
// ---------------------------------------------------------------------------
__global__ void __launch_bounds__(256) build_biasmask(
    const float* __restrict__ mask, const int* __restrict__ rpi,
    const float* __restrict__ rpb, float* __restrict__ BM) {
  const int wim  = blockIdx.x;   // 0..255
  const int head = blockIdx.y;   // 0..11
  float* out = BM + ((size_t)(wim * kNH + head)) * 4096;
  for (int i = threadIdx.x; i < 4096; i += 256) {
    const int row = i >> 6, col = i & 63;
    float v = -1e30f;
    if (row < 49 && col < 49)
      v = rpb[rpi[row * 49 + col] * kNH + head] +
          mask[(size_t)wim * 2401 + row * 49 + col];
    out[i] = v;
  }
}

// ---------------------------------------------------------------------------
// Generic WMMA GEMM: out[M,Ncol] = epilogue( A[M,K](bf16) @ W[Ncol,K]^T + bias )
// One wave per 32x64 tile of C: 2 A-fragments x 4 B-fragments, 8 WMMA / k-step.
// ---------------------------------------------------------------------------
template<bool OBF, bool GELU_, bool RESID>
__global__ void __launch_bounds__(32) gemm_wmma_kernel(
    const bf16* __restrict__ A, const bf16* __restrict__ W,
    const float* __restrict__ bias, void* __restrict__ outp,
    const float* __restrict__ resid, int M, int Ncol, int K, float scale)
{
  const int mt = blockIdx.x;                 // 32-row tile
  const int nb = blockIdx.y;                 // 64-col block
  const int lane = threadIdx.x & 31;
  const int li = lane & 15, half = lane >> 4;
  (void)M;

  v8f acc[2][4];
#pragma unroll
  for (int g = 0; g < 2; ++g)
#pragma unroll
    for (int t = 0; t < 4; ++t)
#pragma unroll
      for (int e = 0; e < 8; ++e) acc[g][t][e] = 0.0f;

  const bf16* A0 = A + (size_t)mt * 32 * K;
  const bf16* A1 = A0 + (size_t)16 * K;
#pragma unroll 4
  for (int k0 = 0; k0 < K; k0 += 32) {
    if (k0 + 32 < K) __builtin_prefetch(A0 + k0 + 32, 0, 1);
    v16bf a0 = frag_a(A0 + k0, K);
    v16bf a1 = frag_a(A1 + k0, K);
#pragma unroll
    for (int t = 0; t < 4; ++t) {
      v16bf b = frag_b(W + (size_t)((nb * 4 + t) * 16) * K + k0, K);
      acc[0][t] = wmma_bf16(a0, b, acc[0][t]);
      acc[1][t] = wmma_bf16(a1, b, acc[1][t]);
    }
  }

#pragma unroll
  for (int g = 0; g < 2; ++g)
#pragma unroll
    for (int t = 0; t < 4; ++t) {
      const int col = (nb * 4 + t) * 16 + li;
      const float bv = bias[col];
#pragma unroll
      for (int r = 0; r < 8; ++r) {
        const int row = mt * 32 + g * 16 + half * 8 + r;
        float v = (acc[g][t][r] + bv) * scale;
        if (GELU_) v = 0.5f * v * (1.0f + erff(v * 0.70710678118654752f));
        const size_t o = (size_t)row * Ncol + col;
        if (RESID) v += resid[o];
        if (OBF) ((bf16*)outp)[o] = (bf16)v;
        else     ((float*)outp)[o] = v;
      }
    }
}

// ---------------------------------------------------------------------------
// Shared LayerNorm helper (128 threads, 3 elements per thread, DIM=384)
// ---------------------------------------------------------------------------
__device__ __forceinline__ void ln_reduce_write(float a0, float a1, float a2,
                                                const float* __restrict__ g,
                                                const float* __restrict__ be,
                                                bf16* __restrict__ L, size_t orow,
                                                int tid, float* red) {
  red[tid] = a0 + a1 + a2;
  __syncthreads();
  for (int off = 64; off > 0; off >>= 1) {
    if (tid < off) red[tid] += red[tid + off];
    __syncthreads();
  }
  const float mean = red[0] * (1.0f / kDIM);
  __syncthreads();
  const float d0 = a0 - mean, d1 = a1 - mean, d2 = a2 - mean;
  red[tid] = d0 * d0 + d1 * d1 + d2 * d2;
  __syncthreads();
  for (int off = 64; off > 0; off >>= 1) {
    if (tid < off) red[tid] += red[tid + off];
    __syncthreads();
  }
  const float rs = rsqrtf(red[0] * (1.0f / kDIM) + 1e-5f);
  __syncthreads();
  L[orow + tid]       = (bf16)(d0 * rs * g[tid]       + be[tid]);
  L[orow + tid + 128] = (bf16)(d1 * rs * g[tid + 128] + be[tid + 128]);
  L[orow + tid + 256] = (bf16)(d2 * rs * g[tid + 256] + be[tid + 256]);
  __syncthreads();
}

// ---------------------------------------------------------------------------
// LN1 + cyclic shift (-3,-3) + window partition, both branches -> bf16
// ---------------------------------------------------------------------------
__global__ void __launch_bounds__(128) ln1_window_kernel(
    const float* __restrict__ in1, const float* __restrict__ in2,
    const float* __restrict__ g1, const float* __restrict__ be1,
    const float* __restrict__ g2, const float* __restrict__ be2,
    bf16* __restrict__ X1, bf16* __restrict__ X2)
{
  const int t = blockIdx.x;
  const int tid = threadIdx.x;
  const int wr  = t % 49;
  const int win = t / 49;
  const int b   = win >> 8;
  const int wim = win & 255;
  int h = (wim >> 4) * 7 + wr / 7 + 3; if (h >= 112) h -= 112;
  int w = (wim & 15) * 7 + wr % 7 + 3; if (w >= 112) w -= 112;
  const size_t irow = ((size_t)b * 12544 + (size_t)h * 112 + w) * kDIM;
  const size_t orow = (size_t)t * kDIM;
  __shared__ float red[128];

  {
    float x0 = in1[irow + tid], x1 = in1[irow + tid + 128], x2 = in1[irow + tid + 256];
    ln_reduce_write(x0, x1, x2, g1, be1, X1, orow, tid, red);
  }
  {
    float x0 = in2[irow + tid], x1 = in2[irow + tid + 128], x2 = in2[irow + tid + 256];
    ln_reduce_write(x0, x1, x2, g2, be2, X2, orow, tid, red);
  }
}

// ---------------------------------------------------------------------------
// Windowed cross-attention for one (window, head).
// 512 threads = 16 waves. LDS-staged zero-padded tiles; branch-free softmax.
// ---------------------------------------------------------------------------
__global__ void __launch_bounds__(512) attn_kernel(
    const bf16* __restrict__ Q2, const bf16* __restrict__ KV1,
    const float* __restrict__ BM, bf16* __restrict__ O)
{
  const int win  = blockIdx.x;       // 0..1023
  const int head = blockIdx.y;       // 0..11
  const int tid  = threadIdx.x;
  const int wv   = tid >> 5;
  const int lane = tid & 31;
  const int li = lane & 15, half = lane >> 4;

  __shared__ bf16  Qt[64][32];       // q rows (padded to 64)
  __shared__ bf16  Kt[64][32];       // k rows
  __shared__ bf16  Vt[32][64];       // v TRANSPOSED: Vt[d][token]
  __shared__ float Ssm[64][65];
  __shared__ bf16  Psm[64][64];

  // ---- stage Q/K/V tiles into LDS, zero-padded (8B per thread per tile) ----
  {
    const int row = tid >> 3;        // 0..63
    const int cg  = tid & 7;         // 4-element column group
    v4bf q, k, v;
#pragma unroll
    for (int j = 0; j < 4; ++j) { q[j] = (bf16)0.0f; k[j] = (bf16)0.0f; v[j] = (bf16)0.0f; }
    if (row < 49) {
      const size_t gq = ((size_t)(win * 49 + row)) * kDIM + head * kHD + cg * 4;
      const size_t gk = ((size_t)(win * 49 + row)) * 768  + head * kHD + cg * 4;
      q = *(const v4bf*)(Q2 + gq);
      k = *(const v4bf*)(KV1 + gk);
      v = *(const v4bf*)(KV1 + gk + 384);
    }
    *(v4bf*)&Qt[row][cg * 4] = q;
    *(v4bf*)&Kt[row][cg * 4] = k;
#pragma unroll
    for (int j = 0; j < 4; ++j) Vt[cg * 4 + j][row] = v[j];
  }
  __syncthreads();

  // ---- phase 1: S = q@k^T + biasmask; 16 waves, one 16x16 WMMA each -------
  {
    const int mi = wv >> 2, ni = wv & 3;
    v16bf a = frag_a(&Qt[mi * 16][0], 32);
    v16bf b = frag_b(&Kt[ni * 16][0], 32);
    v8f acc;
#pragma unroll
    for (int e = 0; e < 8; ++e) acc[e] = 0.0f;
    acc = wmma_bf16(a, b, acc);
    const float* bmp = BM + ((size_t)((win & 255) * kNH + head)) * 4096;
#pragma unroll
    for (int r = 0; r < 8; ++r) {
      const int row = mi * 16 + half * 8 + r;
      const int col = ni * 16 + li;
      Ssm[row][col] = acc[r] + bmp[row * 64 + col];
    }
  }
  __syncthreads();

  // ---- phase 2: branch-free softmax, 8 lanes per row ----------------------
  {
    const int row = tid >> 3;        // 0..63
    const int sub = tid & 7;
    float vals[8];
    float mx = -1e30f;
#pragma unroll
    for (int j = 0; j < 8; ++j) {
      vals[j] = Ssm[row][sub * 8 + j];
      mx = fmaxf(mx, vals[j]);
    }
#pragma unroll
    for (int o = 4; o > 0; o >>= 1) mx = fmaxf(mx, __shfl_xor(mx, o, 8));
    float sum = 0.0f;
#pragma unroll
    for (int j = 0; j < 8; ++j) { vals[j] = __expf(vals[j] - mx); sum += vals[j]; }
#pragma unroll
    for (int o = 4; o > 0; o >>= 1) sum += __shfl_xor(sum, o, 8);
    const float inv = (row < 49) ? 1.0f / sum : 0.0f;   // padded rows -> P = 0
#pragma unroll
    for (int j = 0; j < 8; ++j) Psm[row][sub * 8 + j] = (bf16)(vals[j] * inv);
  }
  __syncthreads();

  // ---- phase 3: O = P @ V; 8 waves, K=64 -> 2 WMMAs each ------------------
  if (wv < 8) {
    const int mi = wv >> 1, ni = wv & 1;
    v8f acc;
#pragma unroll
    for (int e = 0; e < 8; ++e) acc[e] = 0.0f;
#pragma unroll
    for (int k0 = 0; k0 < 64; k0 += 32) {
      v16bf a = frag_a(&Psm[mi * 16][k0], 64);
      v16bf b = frag_b(&Vt[ni * 16][k0], 64);   // Vt[d][token]: K contiguous
      acc = wmma_bf16(a, b, acc);
    }
#pragma unroll
    for (int r = 0; r < 8; ++r) {
      const int row = mi * 16 + half * 8 + r;
      if (row < 49)
        O[((size_t)(win * 49 + row)) * kDIM + head * kHD + ni * 16 + li] =
            (bf16)acc[r];
    }
  }
}

// ---------------------------------------------------------------------------
// Window-reverse + roll(+3,+3) + residual add (both branches) + LN2 -> bf16
// ---------------------------------------------------------------------------
__global__ void __launch_bounds__(128) unshift_add_ln_kernel(
    const float* __restrict__ Y, const float* __restrict__ s1,
    const float* __restrict__ s2,
    const float* __restrict__ g1, const float* __restrict__ be1,
    const float* __restrict__ g2, const float* __restrict__ be2,
    float* __restrict__ B1, float* __restrict__ B2,
    bf16* __restrict__ L1, bf16* __restrict__ L2)
{
  const int r = blockIdx.x;
  const int tid = threadIdx.x;
  const int b   = r / 12544;
  const int rem = r % 12544;
  const int hh  = rem / 112;
  const int ww  = rem % 112;
  int hs = hh - 3; if (hs < 0) hs += 112;
  int ws = ww - 3; if (ws < 0) ws += 112;
  const int win = b * 256 + (hs / 7) * 16 + (ws / 7);
  const int wr  = (hs % 7) * 7 + (ws % 7);
  const size_t trow = ((size_t)win * 49 + wr) * kDIM;
  const size_t orow = (size_t)r * kDIM;
  __shared__ float red[128];

  float a0, a1, a2, c0, c1, c2;
  {
    const int e0 = tid, e1 = tid + 128, e2 = tid + 256;
    const float y0 = Y[trow + e0], y1 = Y[trow + e1], y2 = Y[trow + e2];
    a0 = s1[orow + e0] + y0; a1 = s1[orow + e1] + y1; a2 = s1[orow + e2] + y2;
    c0 = s2[orow + e0] + y0; c1 = s2[orow + e1] + y1; c2 = s2[orow + e2] + y2;
    B1[orow + e0] = a0; B1[orow + e1] = a1; B1[orow + e2] = a2;
    B2[orow + e0] = c0; B2[orow + e1] = c1; B2[orow + e2] = c2;
  }
  ln_reduce_write(a0, a1, a2, g1, be1, L1, orow, tid, red);
  ln_reduce_write(c0, c1, c2, g2, be2, L2, orow, tid, red);
}

// ---------------------------------------------------------------------------
// Host-side orchestration
// ---------------------------------------------------------------------------
extern "C" void kernel_launch(void* const* d_in, const int* in_sizes, int n_in,
                              void* d_out, int out_size, void* d_ws, size_t ws_size,
                              hipStream_t stream) {
  (void)in_sizes; (void)n_in; (void)out_size; (void)ws_size;

  const float* branch1 = (const float*)d_in[0];
  const float* branch2 = (const float*)d_in[1];
  const float* maskm   = (const float*)d_in[2];
  const int*   rpi     = (const int*)  d_in[3];
  const float* n1w1 = (const float*)d_in[4];
  const float* n1b1 = (const float*)d_in[5];
  const float* n1w2 = (const float*)d_in[6];
  const float* n1b2 = (const float*)d_in[7];
  const float* qw1  = (const float*)d_in[8];
  const float* qb1  = (const float*)d_in[9];
  const float* qw2  = (const float*)d_in[10];
  const float* qb2  = (const float*)d_in[11];
  const float* rpb1 = (const float*)d_in[12];
  const float* pw1  = (const float*)d_in[14];
  const float* pb1  = (const float*)d_in[15];
  const float* n2w1 = (const float*)d_in[18];
  const float* n2b1 = (const float*)d_in[19];
  const float* n2w2 = (const float*)d_in[20];
  const float* n2b2 = (const float*)d_in[21];
  const float* m1fc1w = (const float*)d_in[22];
  const float* m1fc1b = (const float*)d_in[23];
  const float* m1fc2w = (const float*)d_in[24];
  const float* m1fc2b = (const float*)d_in[25];
  const float* m2fc1w = (const float*)d_in[26];
  const float* m2fc1b = (const float*)d_in[27];
  const float* m2fc2w = (const float*)d_in[28];
  const float* m2fc2b = (const float*)d_in[29];

  char* ws = (char*)d_ws;
  const size_t MB = (size_t)1 << 20;

  // bf16 weights (persistent): 8 MB region
  bf16* W1kv = (bf16*)(ws + 0);            // qw1 rows 384..1151  (768 x 384)
  bf16* W2q  = W1kv + 768 * 384;           // qw2 rows 0..383     (384 x 384)
  bf16* Pw   = W2q  + 384 * 384;           // pw1                 (384 x 384)
  bf16* F1a  = Pw   + 384 * 384;           // m1fc1w              (1536 x 384)
  bf16* F2a  = F1a  + 1536 * 384;          // m1fc2w              (384 x 1536)
  bf16* F1b  = F2a  + 384 * 1536;          // m2fc1w
  bf16* F2b  = F1b  + 1536 * 384;          // m2fc2w

  bf16*  X1   = (bf16*)(ws + 8 * MB);      // 50176x384 bf16, later aliased by Obuf
  bf16*  X2   = (bf16*)(ws + 48 * MB);     // 50176x384 bf16
  bf16*  KV1  = (bf16*)(ws + 88 * MB);     // 50176x768 bf16, later aliased by Y(f32)
  bf16*  Q2   = (bf16*)(ws + 168 * MB);    // 50176x384 bf16 (pre-scaled)
  float* B1   = (float*)(ws + 208 * MB);   // 50176x384 f32 residual
  float* B2   = (float*)(ws + 288 * MB);
  bf16*  L1   = (bf16*)(ws + 368 * MB);    // LN2 outputs, bf16
  bf16*  L2   = (bf16*)(ws + 408 * MB);
  bf16*  Hbuf = (bf16*)(ws + 448 * MB);    // 50176x1536 bf16 (reused per branch)
  float* BM   = (float*)(ws + 608 * MB);   // 256x12x64x64 f32 bias+mask table
  bf16*  Obuf = X1;                        // attention output reuses X1
  float* Yb   = (float*)KV1;               // projection output reuses KV1

  float* out1 = (float*)d_out;
  float* out2 = out1 + (size_t)kNTOK * kDIM;

  auto cvt = [&](const float* in, bf16* out, int n) {
    const int n4 = n / 4;
    cvt_f32_bf16<<<(n4 + 255) / 256, 256, 0, stream>>>(in, out, n4);
  };
  cvt(qw1 + 384 * 384, W1kv, 768 * 384);   // k1,v1 weights only
  cvt(qw2,             W2q,  384 * 384);   // q2 weights only
  cvt(pw1,             Pw,   384 * 384);
  cvt(m1fc1w, F1a, 1536 * 384);
  cvt(m1fc2w, F2a, 384 * 1536);
  cvt(m2fc1w, F1b, 1536 * 384);
  cvt(m2fc2w, F2b, 384 * 1536);

  // padded bias+mask table
  build_biasmask<<<dim3(256, kNH), 256, 0, stream>>>(maskm, rpi, rpb1, BM);

  // LN1 + shift + window partition
  ln1_window_kernel<<<kNTOK, 128, 0, stream>>>(branch1, branch2,
                                               n1w1, n1b1, n1w2, n1b2, X1, X2);

  // KV1 = X1 @ W1kv^T + bias(k,v)        (bf16 out)
  gemm_wmma_kernel<true, false, false><<<dim3(kNTOK / 32, 768 / 64), 32, 0, stream>>>(
      X1, W1kv, qb1 + 384, KV1, nullptr, kNTOK, 768, 384, 1.0f);
  // Q2  = (X2 @ W2q^T + bias(q)) * SCALE (bf16 out)
  gemm_wmma_kernel<true, false, false><<<dim3(kNTOK / 32, 384 / 64), 32, 0, stream>>>(
      X2, W2q, qb2, Q2, nullptr, kNTOK, 384, 384, kSCALE);

  // attention: (1024 windows x 12 heads)
  attn_kernel<<<dim3(1024, kNH), 512, 0, stream>>>(Q2, KV1, BM, Obuf);

  // projection: Y = O @ pw1^T + pb1 (f32 out, windowed layout)
  gemm_wmma_kernel<false, false, false><<<dim3(kNTOK / 32, 384 / 64), 32, 0, stream>>>(
      Obuf, Pw, pb1, Yb, nullptr, kNTOK, 384, 384, 1.0f);

  // unshift + residual adds + LN2 (both branches share y1)
  unshift_add_ln_kernel<<<kNTOK, 128, 0, stream>>>(Yb, branch1, branch2,
                                                   n2w1, n2b1, n2w2, n2b2,
                                                   B1, B2, L1, L2);

  // MLP branch 1: H = gelu(L1 @ fc1^T + b); out1 = H @ fc2^T + b + B1
  gemm_wmma_kernel<true, true, false><<<dim3(kNTOK / 32, kHID / 64), 32, 0, stream>>>(
      L1, F1a, m1fc1b, Hbuf, nullptr, kNTOK, kHID, 384, 1.0f);
  gemm_wmma_kernel<false, false, true><<<dim3(kNTOK / 32, 384 / 64), 32, 0, stream>>>(
      Hbuf, F2a, m1fc2b, out1, B1, kNTOK, 384, kHID, 1.0f);

  // MLP branch 2
  gemm_wmma_kernel<true, true, false><<<dim3(kNTOK / 32, kHID / 64), 32, 0, stream>>>(
      L2, F1b, m2fc1b, Hbuf, nullptr, kNTOK, kHID, 384, 1.0f);
  gemm_wmma_kernel<false, false, true><<<dim3(kNTOK / 32, 384 / 64), 32, 0, stream>>>(
      Hbuf, F2b, m2fc2b, out2, B2, kNTOK, 384, kHID, 1.0f);
}
